// CrossAttention_46471546143211
// MI455X (gfx1250) — matrix operand order repaired and checked by reading, not compiled, not run on hardware
//
#include <hip/hip_runtime.h>

#define DIMC 192
#define NHEADS 6
#define BATCH 8
#define HH 112
#define WW 112
#define HWSZ (HH * WW)
#define NPIX (BATCH * HWSZ)

#if defined(__has_builtin)
#if __has_builtin(__builtin_amdgcn_global_load_async_to_lds_b128)
#define HAS_ASYNC_LDS 1
#endif
#endif

typedef __attribute__((ext_vector_type(16))) __bf16 v16bf;
typedef __attribute__((ext_vector_type(8)))  __bf16 v8bf;
typedef __attribute__((ext_vector_type(8)))  float  v8f;

// A-matrix fragment layout (16-bit, 16x32): lanes 0-15 hold K=0..7 & 16..23,
// lanes 16-31 hold K=8..15 & 24..31.  base points at (row, kbase) in LDS.
__device__ __forceinline__ v16bf frag_interleaved(const __bf16* base) {
    v8bf lo = *(const v8bf*)(base);
    v8bf hi = *(const v8bf*)(base + 16);
    v16bf r;
#pragma unroll
    for (int i = 0; i < 8; ++i) { r[i] = lo[i]; r[i + 8] = hi[i]; }
    return r;
}

// Contiguous-k fragment (not AMD's table order; unused here): lanes 0-15 hold K=0..15,
// lanes 16-31 hold K=16..31 for their column; contiguous 16 values.
__device__ __forceinline__ v16bf frag_contig(const __bf16* base) {
    v8bf lo = *(const v8bf*)(base);
    v8bf hi = *(const v8bf*)(base + 8);
    v16bf r;
#pragma unroll
    for (int i = 0; i < 8; ++i) { r[i] = lo[i]; r[i + 8] = hi[i]; }
    return r;
}

__device__ __forceinline__ v8f wmma_bf16(v16bf a, v16bf b, v8f c) {
    return __builtin_amdgcn_wmma_f32_16x16x32_bf16(
        /*neg_a=*/false, a, /*neg_b=*/false, b,
        /*c_mod=*/(short)0, c, /*reuse_a=*/false, /*reuse_b=*/false);
}

#ifdef HAS_ASYNC_LDS
// Builtin prototype (from hipcc diagnostic): param 1 is AS(1) int4* (non-const),
// param 2 is the LDS-side pointer.  Cast through uintptr_t to strip const and
// retag address spaces; low 32 bits of a generic LDS pointer are the LDS offset.
typedef int v4i __attribute__((vector_size(16)));
typedef __attribute__((address_space(1))) v4i* as1_v4i_ptr;
typedef __attribute__((address_space(3))) v4i* as3_v4i_ptr;

__device__ __forceinline__ void async_copy_b128(const float* g, float* lds) {
    __builtin_amdgcn_global_load_async_to_lds_b128(
        (as1_v4i_ptr)(uintptr_t)g, (as3_v4i_ptr)(uintptr_t)lds, 0, 0);
}
__device__ __forceinline__ void wait_asynccnt0() {
#if __has_builtin(__builtin_amdgcn_s_wait_asynccnt)
    __builtin_amdgcn_s_wait_asynccnt(0);
#else
    asm volatile("s_wait_asynccnt 0x0" ::: "memory");
#endif
}
#endif

// ---------------------------------------------------------------------------
// conv1x1 as GEMM:  out[p, o] = sum_c in[p, c] * W[o, c] + bias[o]
// IN_NCHW: input laid out NCHW (channel stride = HWSZ), else NHWC (stride 1).
// OUT_NCHW: output NCHW, else NHWC.
// Block: 256 threads (8 waves), M-tile = 128 pixels, wave owns 16 rows.
// A-fragments are loaded once and reused across all output-channel tiles.
// ---------------------------------------------------------------------------
template <int IN_NCHW, int OUT_NCHW>
__global__ __launch_bounds__(256)
void conv1x1_gemm(const float* __restrict__ in, const float* __restrict__ wgt,
                  const float* __restrict__ bias, float* __restrict__ out,
                  int Cout) {
    __shared__ __bf16 Alds[128 * 200];  // 128 rows x 192 K (stride 200, 16B-aligned rows)
    __shared__ __bf16 Wlds[16 * 200];   // 16 out-channels x 192 K
    const int t    = threadIdx.x;
    const int lane = t & 31, wave = t >> 5;
    const int lm = lane & 15, lh = lane >> 4, kb = lh * 8;
    const int rowBlock = blockIdx.x * 128;

    // Stage A tile (f32 -> bf16).  2 threads per row, 96 channels each.
    {
        int row = t >> 1;
        int c0  = (t & 1) * 96;
        int p   = rowBlock + row;
        int b   = p / HWSZ, hw = p - b * HWSZ;
#pragma unroll 4
        for (int c = c0; c < c0 + 96; ++c) {
            float v = IN_NCHW ? in[(b * DIMC + c) * HWSZ + hw]
                              : in[p * DIMC + c];
            Alds[row * 200 + c] = (__bf16)v;
        }
    }
    __syncthreads();

    // Hoisted A fragments: invariant across the nb loop (48 VGPRs).
    v16bf afr[6];
#pragma unroll
    for (int ks = 0; ks < 6; ++ks)
        afr[ks] = frag_interleaved(&Alds[(wave * 16 + lm) * 200 + ks * 32 + kb]);

    for (int nb = 0; nb < Cout; nb += 16) {
        for (int i = t; i < 16 * DIMC; i += 256) {
            int n = i / DIMC, c = i - n * DIMC;
            Wlds[n * 200 + c] = (__bf16)wgt[(nb + n) * DIMC + c];
        }
        if (nb + 16 < Cout) __builtin_prefetch(&wgt[(nb + 16) * DIMC + t], 0, 0);
        __syncthreads();

        v8f acc;
        float bv = bias[nb + lm];
#pragma unroll
        for (int r = 0; r < 8; ++r) acc[r] = bv;

#pragma unroll
        for (int ks = 0; ks < 6; ++ks) {
            v16bf b = frag_interleaved(&Wlds[lm * 200 + ks * 32 + kb]);
            acc = wmma_bf16(afr[ks], b, acc);
        }

        const int colg = nb + lm;
#pragma unroll
        for (int r = 0; r < 8; ++r) {
            int row = rowBlock + wave * 16 + r + (lh << 3);
            if (OUT_NCHW) {
                int b2 = row / HWSZ, hw = row - b2 * HWSZ;
                out[(size_t)(b2 * Cout + colg) * HWSZ + hw] = acc[r];
            } else {
                out[(size_t)row * Cout + colg] = acc[r];
            }
        }
        __syncthreads();  // Wlds readers done before next staging
    }
}

// ---------------------------------------------------------------------------
// Relative-position bias MLP: (N^2, 2) -> relu(256) -> 6 heads.
// out layout: [head][n][m], head slices padded to headStride (16B multiple).
// ---------------------------------------------------------------------------
__global__ void rel_bias_mlp(const float* __restrict__ w1, const float* __restrict__ b1,
                             const float* __restrict__ w2, const float* __restrict__ b2,
                             float* __restrict__ out, int ws, int headStride) {
    int N   = ws * ws;
    int tot = N * N;
    int idx = blockIdx.x * blockDim.x + threadIdx.x;
    if (idx >= tot) return;
    int n = idx / N, m = idx - n * N;
    float dy = (float)(n / ws - m / ws);
    float dx = (float)(n % ws - m % ws);
    float ry = copysignf(log1pf(fabsf(dy)), dy);
    float rx = copysignf(log1pf(fabsf(dx)), dx);
    float acc[NHEADS];
#pragma unroll
    for (int h = 0; h < NHEADS; ++h) acc[h] = b2[h];
    for (int j = 0; j < 256; ++j) {
        float hv = fmaxf(0.f, ry * w1[2 * j] + rx * w1[2 * j + 1] + b1[j]);
#pragma unroll
        for (int h = 0; h < NHEADS; ++h) acc[h] += hv * w2[h * 256 + j];
    }
#pragma unroll
    for (int h = 0; h < NHEADS; ++h) out[h * headStride + n * N + m] = acc[h];
}

// ---------------------------------------------------------------------------
// Window attention: one block per (window, head), 4 waves, window padded to 64.
// qk: NHWC P x 384 (q = ch 0..191, k = ch 192..383); v: NHWC P x 192.
// out: NHWC P x 192 (f32).  Bias table staged via async global->LDS copies.
// ---------------------------------------------------------------------------
template <int WS>
__global__ __launch_bounds__(128)
void win_attn(const float* __restrict__ qk, const float* __restrict__ vbuf,
              const float* __restrict__ bias, float* __restrict__ outbuf) {
    constexpr int N   = WS * WS;                // 64 or 49
    constexpr int NP  = 64;                     // padded window size
    constexpr int WG  = 112 / WS;               // windows per axis
    constexpr int NN  = N * N;
    constexpr int NN4 = ((NN + 3) / 4) * 4;     // head stride (16B multiple)

    __shared__ __bf16 qs [NP * 40];             // q,  row stride 40 (16B aligned)
    __shared__ __bf16 ksm[NP * 40];             // k
    __shared__ __bf16 vT [32 * 72];             // v transposed: [d][m]
    __shared__ __bf16 at [NP * 72];             // softmax(attn) in bf16: [n][m]
    __shared__ alignas(16) float biasLds[4096]; // per-head bias slice (<= 64*64)

    const int t    = threadIdx.x;
    const int head = blockIdx.x % NHEADS;
    int wi         = blockIdx.x / NHEADS;
    const int wc = wi % WG; wi /= WG;
    const int wr = wi % WG;
    const int b  = wi / WG;
    const float scale = 0.17677669529663687f;   // 32^-0.5

    // ---- bias slice: async DMA global -> LDS (ASYNCcnt), overlapped with q/k/v staging
    const float* bsrc = bias + head * NN4;
#ifdef HAS_ASYNC_LDS
    {
        constexpr int NQ = NN / 4;              // full 16B quads
        for (int q = t; q < NQ; q += 128)
            async_copy_b128(bsrc + q * 4, &biasLds[q * 4]);
        for (int i = NQ * 4 + t; i < NN; i += 128)  // tail (ws=7: 1 element)
            biasLds[i] = bsrc[i];
    }
#else
    for (int i = t; i < NN; i += 128) biasLds[i] = bsrc[i];
#endif

    // ---- q/k/v staging (f32 -> bf16), v stored transposed
    for (int idx = t; idx < NP * 32; idx += 128) {
        int n = idx >> 5, d = idx & 31;
        float qv = 0.f, kv = 0.f, vv = 0.f;
        if (n < N) {
            int rr = n / WS, cc = n % WS;
            int p  = b * HWSZ + (wr * WS + rr) * WW + (wc * WS + cc);
            qv = qk[p * 384 + head * 32 + d] * scale;
            kv = qk[p * 384 + 192 + head * 32 + d];
            vv = vbuf[p * 192 + head * 32 + d];
        }
        qs [n * 40 + d] = (__bf16)qv;
        ksm[n * 40 + d] = (__bf16)kv;
        vT [d * 72 + n] = (__bf16)vv;
    }
#ifdef HAS_ASYNC_LDS
    wait_asynccnt0();
#endif
    __syncthreads();

    const int lane = t & 31, wave = t >> 5;
    const int lm = lane & 15, lh = lane >> 4, kb = lh * 8;
    const int rowBase = wave * 16;

    // attn = q @ k^T + bias   (K = 32, single WMMA per 16x16 tile)
    v16bf qa = frag_interleaved(&qs[(rowBase + lm) * 40 + kb]);
    v8f acc[4];
#pragma unroll
    for (int tn = 0; tn < 4; ++tn) {
        int m = tn * 16 + lm;
#pragma unroll
        for (int r = 0; r < 8; ++r) {
            int n = rowBase + r + (lh << 3);
            float bv = -1e30f;  // mask padded key columns
            if (m < N) bv = (n < N) ? biasLds[n * N + m] : 0.f;
            acc[tn][r] = bv;
        }
        v16bf kf = frag_interleaved(&ksm[m * 40 + kb]);
        acc[tn] = wmma_bf16(qa, kf, acc[tn]);
    }

    // row-wise softmax over 64 keys (4 tiles x 16-lane reduction)
#pragma unroll
    for (int r = 0; r < 8; ++r) {
        float x0 = acc[0][r], x1 = acc[1][r], x2 = acc[2][r], x3 = acc[3][r];
        float mx = fmaxf(fmaxf(x0, x1), fmaxf(x2, x3));
#pragma unroll
        for (int off = 1; off < 16; off <<= 1) mx = fmaxf(mx, __shfl_xor(mx, off, 32));
        float e0 = __expf(x0 - mx), e1 = __expf(x1 - mx);
        float e2 = __expf(x2 - mx), e3 = __expf(x3 - mx);
        float sm = e0 + e1 + e2 + e3;
#pragma unroll
        for (int off = 1; off < 16; off <<= 1) sm += __shfl_xor(sm, off, 32);
        float inv = 1.0f / sm;
        int n = rowBase + r + (lh << 3);
        at[n * 72 +  0 + lm] = (__bf16)(e0 * inv);
        at[n * 72 + 16 + lm] = (__bf16)(e1 * inv);
        at[n * 72 + 32 + lm] = (__bf16)(e2 * inv);
        at[n * 72 + 48 + lm] = (__bf16)(e3 * inv);
    }
    __syncthreads();

    // out = attn @ v   (K = 64 -> 2 WMMA steps, N = 32 -> 2 tiles)
    v8f oacc[2];
#pragma unroll
    for (int dt = 0; dt < 2; ++dt)
#pragma unroll
        for (int r = 0; r < 8; ++r) oacc[dt][r] = 0.f;
#pragma unroll
    for (int ks2 = 0; ks2 < 2; ++ks2) {
        v16bf af = frag_interleaved(&at[(rowBase + lm) * 72 + ks2 * 32 + kb]);
#pragma unroll
        for (int dt = 0; dt < 2; ++dt) {
            v16bf vf = frag_interleaved(&vT[(dt * 16 + lm) * 72 + ks2 * 32 + kb]);
            oacc[dt] = wmma_bf16(af, vf, oacc[dt]);
        }
    }

#pragma unroll
    for (int dt = 0; dt < 2; ++dt)
#pragma unroll
        for (int r = 0; r < 8; ++r) {
            int n = rowBase + r + (lh << 3);
            if (n < N) {
                int rr = n / WS, cc = n % WS;
                int p  = b * HWSZ + (wr * WS + rr) * WW + (wc * WS + cc);
                outbuf[p * 192 + head * 32 + dt * 16 + lm] = oacc[dt][r];
            }
        }
}

// ---------------------------------------------------------------------------
// Depthwise 5x5 conv with reflect padding + add attention output (all NHWC).
// ---------------------------------------------------------------------------
__global__ __launch_bounds__(256)
void dwconv5_add(const float* __restrict__ vin, const float* __restrict__ cw,
                 const float* __restrict__ cb, const float* __restrict__ attn,
                 float* __restrict__ out) {
    int gid = blockIdx.x * 256 + threadIdx.x;
    if (gid >= NPIX * DIMC) return;
    int c  = gid % DIMC;
    int p  = gid / DIMC;
    int b  = p / HWSZ, hw = p - b * HWSZ;
    int h  = hw / WW, w0 = hw - h * WW;
    float s = cb[c];
#pragma unroll
    for (int dy = 0; dy < 5; ++dy) {
        int ih = h + dy - 2;
        ih = ih < 0 ? -ih : ih;
        ih = ih >= HH ? 2 * HH - 2 - ih : ih;
#pragma unroll
        for (int dx = 0; dx < 5; ++dx) {
            int iw = w0 + dx - 2;
            iw = iw < 0 ? -iw : iw;
            iw = iw >= WW ? 2 * WW - 2 - iw : iw;
            s += vin[(size_t)(b * HWSZ + ih * WW + iw) * DIMC + c] * cw[c * 25 + dy * 5 + dx];
        }
    }
    out[gid] = s + attn[gid];
}

// ---------------------------------------------------------------------------

extern "C" void kernel_launch(void* const* d_in, const int* in_sizes, int n_in,
                              void* d_out, int out_size, void* d_ws, size_t ws_size,
                              hipStream_t stream) {
    const float* X       = (const float*)d_in[0];
    const float* Y       = (const float*)d_in[1];
    const float* Vm_w    = (const float*)d_in[2];
    const float* Vm_b    = (const float*)d_in[3];
    const float* Vs_w    = (const float*)d_in[4];
    const float* Vs_b    = (const float*)d_in[5];
    const float* QKm_w   = (const float*)d_in[6];
    const float* QKm_b   = (const float*)d_in[7];
    const float* QKs_w   = (const float*)d_in[8];
    const float* QKs_b   = (const float*)d_in[9];
    const float* convm_w = (const float*)d_in[10];
    const float* convm_b = (const float*)d_in[11];
    const float* convs_w = (const float*)d_in[12];
    const float* convs_b = (const float*)d_in[13];
    const float* proj_w  = (const float*)d_in[14];
    const float* proj_b  = (const float*)d_in[15];
    const float* mm_w1   = (const float*)d_in[16];
    const float* mm_b1   = (const float*)d_in[17];
    const float* mm_w2   = (const float*)d_in[18];
    const float* mm_b2   = (const float*)d_in[19];
    const float* ms_w1   = (const float*)d_in[20];
    const float* ms_b1   = (const float*)d_in[21];
    const float* ms_w2   = (const float*)d_in[22];
    const float* ms_b2   = (const float*)d_in[23];

    float* ws = (float*)d_ws;
    const int strideBm = 64 * 64;                    // 4096, already 16B multiple
    const int strideBs = ((49 * 49 + 3) / 4) * 4;    // 2404
    size_t oVm  = 0;
    size_t oVs  = oVm  + (size_t)NPIX * DIMC;
    size_t oQKm = oVs  + (size_t)NPIX * DIMC;
    size_t oQKs = oQKm + (size_t)NPIX * 2 * DIMC;
    size_t oAm  = oQKs + (size_t)NPIX * 2 * DIMC;
    size_t oAs  = oAm  + (size_t)NPIX * DIMC;
    size_t oBm  = oAs  + (size_t)NPIX * DIMC;
    size_t oBs  = oBm  + (size_t)(NHEADS * strideBm);

    float* mainOut   = (float*)d_out;
    float* structOut = mainOut + (size_t)NPIX * DIMC;

    const int gemmGrid = NPIX / 128;  // 784

    // Front conv1x1 GEMMs (NCHW in -> NHWC out, bf16 WMMA)
    conv1x1_gemm<1, 0><<<gemmGrid, 256, 0, stream>>>(X, Vm_w,  Vm_b,  ws + oVm,  DIMC);
    conv1x1_gemm<1, 0><<<gemmGrid, 256, 0, stream>>>(Y, Vs_w,  Vs_b,  ws + oVs,  DIMC);
    conv1x1_gemm<1, 0><<<gemmGrid, 256, 0, stream>>>(X, QKm_w, QKm_b, ws + oQKm, 2 * DIMC);
    conv1x1_gemm<1, 0><<<gemmGrid, 256, 0, stream>>>(Y, QKs_w, QKs_b, ws + oQKs, 2 * DIMC);

    // Relative-position bias tables (head slices padded to 16B multiples)
    rel_bias_mlp<<<(64 * 64 + 255) / 256, 256, 0, stream>>>(mm_w1, mm_b1, mm_w2, mm_b2,
                                                            ws + oBm, 8, strideBm);
    rel_bias_mlp<<<(49 * 49 + 255) / 256, 256, 0, stream>>>(ms_w1, ms_b1, ms_w2, ms_b2,
                                                            ws + oBs, 7, strideBs);

    // Window attention (M: 8x8 windows with v=V_s; S: 7x7 windows with v=V_m)
    win_attn<8><<<8 * 14 * 14 * NHEADS, 128, 0, stream>>>(ws + oQKm, ws + oVs, ws + oBm, ws + oAm);
    win_attn<7><<<8 * 16 * 16 * NHEADS, 128, 0, stream>>>(ws + oQKs, ws + oVm, ws + oBs, ws + oAs);

    // Depthwise conv + attention add; results reuse retired QK workspace.
    const int dwGrid = (NPIX * DIMC + 255) / 256;
    dwconv5_add<<<dwGrid, 256, 0, stream>>>(ws + oVm, convm_w, convm_b, ws + oAm, ws + oQKm);
    dwconv5_add<<<dwGrid, 256, 0, stream>>>(ws + oVs, convs_w, convs_b, ws + oAs, ws + oQKs);

    // Final projection (NHWC in -> NCHW out, shared weights)
    conv1x1_gemm<0, 1><<<gemmGrid, 256, 0, stream>>>(ws + oQKm, proj_w, proj_b, mainOut,   DIMC);
    conv1x1_gemm<0, 1><<<gemmGrid, 256, 0, stream>>>(ws + oQKs, proj_w, proj_b, structOut, DIMC);
}
